// FAGCNEncoder_88304527606664
// MI455X (gfx1250) — matrix operand order repaired and checked by reading, not compiled, not run on hardware
//
#include <hip/hip_runtime.h>
#include <cstdint>
#include <cstddef>

// ---------------- problem constants (from reference) ----------------
#define N_NODES   50000
#define N_EDGES   800000
#define N_GRAPHS  64
#define IN_DIM    512
#define HID       256
#define OUT_DIM   256
#define EPS_F     0.1f

typedef __attribute__((ext_vector_type(16))) __bf16 v16bf;
typedef __attribute__((ext_vector_type(8)))  __bf16 v8bf;
typedef __attribute__((ext_vector_type(8)))  float  v8f;

__device__ __forceinline__ __bf16 f2bf(float f) {
  // round-to-nearest-even f32 -> bf16
  unsigned u = __builtin_bit_cast(unsigned, f);
  unsigned r = (u + 0x7FFFu + ((u >> 16) & 1u)) >> 16;
  unsigned short s = (unsigned short)r;
  return __builtin_bit_cast(__bf16, s);
}
__device__ __forceinline__ unsigned short f2bfu(float f) {
  unsigned u = __builtin_bit_cast(unsigned, f);
  return (unsigned short)((u + 0x7FFFu + ((u >> 16) & 1u)) >> 16);
}

// ---------------- degree / norm ----------------
__global__ void deg_kernel(const int* __restrict__ dst, float* __restrict__ deg, int E) {
  int i = blockIdx.x * blockDim.x + threadIdx.x;
  if (i < E) atomicAdd(&deg[dst[i]], 1.0f);
}

__global__ void dinv_kernel(float* __restrict__ deg, int n) {
  int i = blockIdx.x * blockDim.x + threadIdx.x;
  if (i < n) {
    float d = deg[i];
    deg[i] = (d > 0.0f) ? rsqrtf(fmaxf(d, 1.0f)) : 0.0f;  // in-place deg -> dinv
  }
}

__global__ void norm_kernel(const int* __restrict__ src, const int* __restrict__ dst,
                            const float* __restrict__ dinv, float* __restrict__ nrm, int E) {
  int i = blockIdx.x * blockDim.x + threadIdx.x;
  if (i < E) nrm[i] = dinv[src[i]] * dinv[dst[i]];
}

// ---------------- f32 -> bf16 bulk convert (straight layout) ----------------
__global__ void f32_to_bf16_kernel(const float* __restrict__ in, __bf16* __restrict__ out,
                                   int total4) {
  int i = blockIdx.x * blockDim.x + threadIdx.x;
  if (i >= total4) return;
  float4 v = ((const float4*)in)[i];
  ushort4 o;
  o.x = f2bfu(v.x); o.y = f2bfu(v.y); o.z = f2bfu(v.z); o.w = f2bfu(v.w);
  ((ushort4*)out)[i] = o;
}

// ---------------- weight transpose + f32->bf16 convert ----------------
// Wt[n*K + k] = bf16(W[k*N + n])
__global__ void convertW_kernel(const float* __restrict__ W, __bf16* __restrict__ Wt,
                                int K, int N) {
  int idx = blockIdx.x * blockDim.x + threadIdx.x;
  if (idx >= K * N) return;
  int k = idx / N;
  int n = idx - k * N;
  Wt[(size_t)n * K + k] = f2bf(W[idx]);
}

// ---------------- WMMA GEMM: C[M,N] = A[M,K](bf16) * Bt[N,K](bf16)^T + bias ----------------
// One wave computes a 16x64 output strip (4 WMMA n-tiles), looping K in steps of 32.
__global__ void gemm_bf16_wmma(const __bf16* __restrict__ A,
                               const __bf16* __restrict__ Bt,
                               const float* __restrict__ bias,
                               float* __restrict__ C,
                               int M, int K, int N) {
  const int lane    = threadIdx.x & 31;
  const int wave    = (blockIdx.x * blockDim.x + threadIdx.x) >> 5;
  const int nGroups = N >> 6;            // 64 columns per wave
  const int mTiles  = M >> 4;
  const int mTile   = wave / nGroups;
  const int g       = wave - mTile * nGroups;
  if (mTile >= mTiles) return;

  const int r    = lane & 15;            // row within tile (A) / column within tile (B)
  const int half = lane >> 4;            // lane half selects K sub-ranges
  const __bf16* aPtr = A + (size_t)(mTile * 16 + r) * K;

  v8f acc[4];
#pragma unroll
  for (int t = 0; t < 4; ++t) {
    float bv = bias[g * 64 + t * 16 + r];
#pragma unroll
    for (int v = 0; v < 8; ++v) acc[t][v] = bv;
  }

  for (int kk = 0; kk < K; kk += 32) {
    // A fragment (half h holds K [h*16, h*16+16), matching B)
    v16bf a = *(const v16bf*)(aPtr + kk + half * 16);

    const int kB = kk + half * 16;       // B layout: half h holds K [h*16, h*16+16)
#pragma unroll
    for (int t = 0; t < 4; ++t) {
      v16bf b = *(const v16bf*)(Bt + (size_t)(g * 64 + t * 16 + r) * K + kB);
      acc[t] = __builtin_amdgcn_wmma_f32_16x16x32_bf16(
          false, a, false, b, (short)0, acc[t], false, false);
    }
  }

  // D layout: VGPR v, lane half h -> row = v + 8*h, col = lane&15
#pragma unroll
  for (int t = 0; t < 4; ++t) {
    int col = g * 64 + t * 16 + r;
#pragma unroll
    for (int v = 0; v < 8; ++v) {
      int row = mTile * 16 + v + half * 8;
      C[(size_t)row * N + col] = acc[t][v];
    }
  }
}

// ---------------- per-node attention scalars: al = h . wl + bl, ar = h . wr + br ----------------
__global__ void att_kernel(const float* __restrict__ h,
                           const float* __restrict__ wl, const float* __restrict__ wr,
                           const float* __restrict__ blp, const float* __restrict__ brp,
                           float* __restrict__ al, float* __restrict__ ar, int n) {
  int node = (blockIdx.x * blockDim.x + threadIdx.x) >> 5;
  int lane = threadIdx.x & 31;
  if (node >= n) return;
  const float4* hp  = (const float4*)(h + (size_t)node * HID);
  const float4* wl4 = (const float4*)wl;
  const float4* wr4 = (const float4*)wr;
  float4 h0 = hp[lane * 2], h1 = hp[lane * 2 + 1];
  float4 l0 = wl4[lane * 2], l1 = wl4[lane * 2 + 1];
  float4 r0 = wr4[lane * 2], r1 = wr4[lane * 2 + 1];
  float sl = h0.x * l0.x + h0.y * l0.y + h0.z * l0.z + h0.w * l0.w
           + h1.x * l1.x + h1.y * l1.y + h1.z * l1.z + h1.w * l1.w;
  float sr = h0.x * r0.x + h0.y * r0.y + h0.z * r0.z + h0.w * r0.w
           + h1.x * r1.x + h1.y * r1.y + h1.z * r1.z + h1.w * r1.w;
#pragma unroll
  for (int off = 16; off > 0; off >>= 1) {
    sl += __shfl_xor(sl, off, 32);
    sr += __shfl_xor(sr, off, 32);
  }
  if (lane == 0) {
    al[node] = sl + blp[0];
    ar[node] = sr + brp[0];
  }
}

// ---------------- edge kernel: agg[dst] += h[src] * (norm * tanh(al[src]+ar[dst])) ----------------
// one wave per edge; each lane handles 8 channels
__global__ void edge_kernel(const int* __restrict__ src, const int* __restrict__ dst,
                            const float* __restrict__ nrm,
                            const float* __restrict__ al, const float* __restrict__ ar,
                            const float* __restrict__ h, float* __restrict__ agg, int E) {
  int e    = (blockIdx.x * blockDim.x + threadIdx.x) >> 5;
  int lane = threadIdx.x & 31;
  if (e >= E) return;
  int s = src[e], d = dst[e];
  float w = nrm[e] * tanhf(al[s] + ar[d]);
  const float4* hp = (const float4*)(h + (size_t)s * HID);
  float* ap = agg + (size_t)d * HID + lane * 8;
  float4 v0 = hp[lane * 2], v1 = hp[lane * 2 + 1];
  atomicAdd(ap + 0, v0.x * w); atomicAdd(ap + 1, v0.y * w);
  atomicAdd(ap + 2, v0.z * w); atomicAdd(ap + 3, v0.w * w);
  atomicAdd(ap + 4, v1.x * w); atomicAdd(ap + 5, v1.y * w);
  atomicAdd(ap + 6, v1.z * w); atomicAdd(ap + 7, v1.w * w);
}

// ---------------- h = EPS*h0 + agg (+ReLU); optional fused bf16 emission ----------------
__global__ void combine_kernel(const float* __restrict__ h0, const float* __restrict__ agg,
                               float* __restrict__ h, __bf16* __restrict__ hbf,
                               int total4, int relu) {
  int i = blockIdx.x * blockDim.x + threadIdx.x;
  if (i >= total4) return;
  float4 a = ((const float4*)h0)[i];
  float4 b = ((const float4*)agg)[i];
  float4 o;
  o.x = EPS_F * a.x + b.x;
  o.y = EPS_F * a.y + b.y;
  o.z = EPS_F * a.z + b.z;
  o.w = EPS_F * a.w + b.w;
  if (relu) {
    o.x = fmaxf(o.x, 0.0f); o.y = fmaxf(o.y, 0.0f);
    o.z = fmaxf(o.z, 0.0f); o.w = fmaxf(o.w, 0.0f);
  }
  ((float4*)h)[i] = o;
  if (hbf) {
    ushort4 ob;
    ob.x = f2bfu(o.x); ob.y = f2bfu(o.y); ob.z = f2bfu(o.z); ob.w = f2bfu(o.w);
    ((ushort4*)hbf)[i] = ob;
  }
}

// ---------------- graph mean pooling ----------------
__global__ void pool_kernel(const float* __restrict__ nr, const int* __restrict__ batch,
                            float* __restrict__ sums, float* __restrict__ cnt, int n) {
  int node = (blockIdx.x * blockDim.x + threadIdx.x) >> 5;
  int lane = threadIdx.x & 31;
  if (node >= n) return;
  int g = batch[node];
  const float4* p = (const float4*)(nr + (size_t)node * OUT_DIM);
  float* sp = sums + (size_t)g * OUT_DIM + lane * 8;
  float4 v0 = p[lane * 2], v1 = p[lane * 2 + 1];
  atomicAdd(sp + 0, v0.x); atomicAdd(sp + 1, v0.y);
  atomicAdd(sp + 2, v0.z); atomicAdd(sp + 3, v0.w);
  atomicAdd(sp + 4, v1.x); atomicAdd(sp + 5, v1.y);
  atomicAdd(sp + 6, v1.z); atomicAdd(sp + 7, v1.w);
  if (lane == 0) atomicAdd(cnt + g, 1.0f);
}

__global__ void pool_div_kernel(const float* __restrict__ sums, const float* __restrict__ cnt,
                                float* __restrict__ out) {
  int i = blockIdx.x * blockDim.x + threadIdx.x;
  if (i >= N_GRAPHS * OUT_DIM) return;
  out[i] = sums[i] / fmaxf(cnt[i >> 8], 1.0f);   // OUT_DIM == 256
}

// ---------------- host orchestration ----------------
static inline int cdiv(long long a, long long b) { return (int)((a + b - 1) / b); }

extern "C" void kernel_launch(void* const* d_in, const int* in_sizes, int n_in,
                              void* d_out, int out_size, void* d_ws, size_t ws_size,
                              hipStream_t stream) {
  (void)in_sizes; (void)n_in; (void)out_size; (void)ws_size;

  const float* x        = (const float*)d_in[0];
  const int*   ei       = (const int*)d_in[1];
  const int*   src      = ei;
  const int*   dst      = ei + N_EDGES;
  const int*   batch    = (const int*)d_in[2];
  const float* W_in     = (const float*)d_in[3];
  const float* b_in     = (const float*)d_in[4];
  const float* att_l_w  = (const float*)d_in[5];
  const float* att_l_b  = (const float*)d_in[6];
  const float* att_r_w  = (const float*)d_in[7];
  const float* att_r_b  = (const float*)d_in[8];
  const float* W_out    = (const float*)d_in[9];
  const float* b_out    = (const float*)d_in[10];

  float* node_repr  = (float*)d_out;                                   // [N_NODES, OUT_DIM]
  float* graph_repr = node_repr + (size_t)N_NODES * OUT_DIM;           // [N_GRAPHS, OUT_DIM]

  // workspace carve-up (256B aligned)
  char* wsp = (char*)d_ws;
  auto alloc = [&](size_t bytes) -> void* {
    void* p = (void*)wsp;
    wsp += (bytes + 255) & ~(size_t)255;
    return p;
  };
  float*  h     = (float*)alloc((size_t)N_NODES * HID * 4);
  float*  h0    = (float*)alloc((size_t)N_NODES * HID * 4);
  float*  agg   = (float*)alloc((size_t)N_NODES * HID * 4);
  __bf16* xbf   = (__bf16*)alloc((size_t)N_NODES * IN_DIM * 2);
  __bf16* hbf   = (__bf16*)alloc((size_t)N_NODES * HID * 2);
  float*  al    = (float*)alloc((size_t)N_NODES * 4);
  float*  ar    = (float*)alloc((size_t)N_NODES * 4);
  float*  dinv  = (float*)alloc((size_t)N_NODES * 4);
  float*  nrm   = (float*)alloc((size_t)N_EDGES * 4);
  __bf16* WtIn  = (__bf16*)alloc((size_t)IN_DIM * HID * 2);
  __bf16* WtOut = (__bf16*)alloc((size_t)HID * OUT_DIM * 2);
  float*  sums  = (float*)alloc((size_t)N_GRAPHS * OUT_DIM * 4);
  float*  cnt   = (float*)alloc((size_t)N_GRAPHS * 4);

  // --- degrees / symmetric norm ---
  hipMemsetAsync(dinv, 0, (size_t)N_NODES * 4, stream);
  deg_kernel<<<cdiv(N_EDGES, 256), 256, 0, stream>>>(dst, dinv, N_EDGES);
  dinv_kernel<<<cdiv(N_NODES, 256), 256, 0, stream>>>(dinv, N_NODES);
  norm_kernel<<<cdiv(N_EDGES, 256), 256, 0, stream>>>(src, dst, dinv, nrm, N_EDGES);

  // --- operand conversion to bf16 ---
  f32_to_bf16_kernel<<<cdiv((long long)N_NODES * IN_DIM / 4, 256), 256, 0, stream>>>(
      x, xbf, N_NODES * IN_DIM / 4);
  convertW_kernel<<<cdiv((long long)IN_DIM * HID, 256), 256, 0, stream>>>(W_in, WtIn, IN_DIM, HID);
  convertW_kernel<<<cdiv((long long)HID * OUT_DIM, 256), 256, 0, stream>>>(W_out, WtOut, HID, OUT_DIM);

  // --- h = x @ W_in + b_in  (WMMA, bf16 operands, f32 accum) ---
  {
    long long waves = (long long)(N_NODES / 16) * (HID / 64);
    gemm_bf16_wmma<<<cdiv(waves * 32, 256), 256, 0, stream>>>(xbf, WtIn, b_in, h,
                                                              N_NODES, IN_DIM, HID);
  }
  hipMemcpyAsync(h0, h, (size_t)N_NODES * HID * 4, hipMemcpyDeviceToDevice, stream);

  // --- 2 FAConv layers ---
  for (int l = 0; l < 2; ++l) {
    att_kernel<<<cdiv((long long)N_NODES * 32, 256), 256, 0, stream>>>(
        h, att_l_w + (size_t)l * HID, att_r_w + (size_t)l * HID,
        att_l_b + l, att_r_b + l, al, ar, N_NODES);
    hipMemsetAsync(agg, 0, (size_t)N_NODES * HID * 4, stream);
    edge_kernel<<<cdiv((long long)N_EDGES * 32, 256), 256, 0, stream>>>(
        src, dst, nrm, al, ar, h, agg, N_EDGES);
    // fuse bf16 emission on the last layer (feeds the output WMMA GEMM)
    combine_kernel<<<cdiv((long long)N_NODES * HID / 4, 256), 256, 0, stream>>>(
        h0, agg, h, (l == 1) ? hbf : (__bf16*)nullptr,
        N_NODES * HID / 4, (l == 0) ? 1 : 0);
  }

  // --- node_repr = h @ W_out + b_out (WMMA), written straight to d_out ---
  {
    long long waves = (long long)(N_NODES / 16) * (OUT_DIM / 64);
    gemm_bf16_wmma<<<cdiv(waves * 32, 256), 256, 0, stream>>>(hbf, WtOut, b_out, node_repr,
                                                              N_NODES, HID, OUT_DIM);
  }

  // --- global mean pool ---
  hipMemsetAsync(sums, 0, (size_t)N_GRAPHS * OUT_DIM * 4, stream);
  hipMemsetAsync(cnt, 0, (size_t)N_GRAPHS * 4, stream);
  pool_kernel<<<cdiv((long long)N_NODES * 32, 256), 256, 0, stream>>>(node_repr, batch, sums, cnt, N_NODES);
  pool_div_kernel<<<cdiv((long long)N_GRAPHS * OUT_DIM, 256), 256, 0, stream>>>(sums, cnt, graph_repr);
}
